// Decoder_27324581937395
// MI455X (gfx1250) — matrix operand order repaired and checked by reading, not compiled, not run on hardware
//
#include <hip/hip_runtime.h>
#include <hip/hip_bf16.h>
#include <math.h>

// ---------------- problem constants ----------------
#define B_    64
#define S_    1024
#define P_    1024
#define DIN_  50
#define D_    300     // true feature dim
#define DP_   320     // padded feature dim (multiple of 32 for WMMA K)
#define OUT_  1024
#define BS_   (B_ * S_)

typedef __bf16 bf16_t;
typedef __attribute__((ext_vector_type(8)))  bf16_t v8bf;
typedef __attribute__((ext_vector_type(16))) bf16_t v16bf;
typedef __attribute__((ext_vector_type(8)))  float  v8f;

// CDNA5 async global->LDS path (guarded: falls back to ld/st staging)
#if defined(__gfx1250__) && __has_builtin(__builtin_amdgcn_global_load_async_to_lds_b128) && __has_builtin(__builtin_amdgcn_s_wait_asynccnt)
#define USE_ASYNC 1
#else
#define USE_ASYNC 0
#endif

typedef int v4i_async __attribute__((vector_size(16)));

__device__ __forceinline__ void cp_async16(const bf16_t* g, bf16_t* l) {
#if USE_ASYNC
  __builtin_amdgcn_global_load_async_to_lds_b128(
      (__attribute__((address_space(1))) v4i_async*)(g),
      (__attribute__((address_space(3))) v4i_async*)(l), 0, 0);
#else
  (void)g; (void)l;
#endif
}

// ======================================================================
// Projection of x: token = LayerNorm(x@ME), past_token = tanh(x@Wpt+b),
// expose_vector = tanh(x@Wex+b).  One block (320 threads) per row.
// ======================================================================
__device__ inline float block_sum_320(float v, float* red, int d) {
  red[d] = v;
  __syncthreads();
  for (int s = 160; s >= 5; s >>= 1) {
    if (d < s) red[d] += red[d + s];
    __syncthreads();
  }
  float r = red[0] + red[1] + red[2] + red[3] + red[4];
  __syncthreads();
  return r;
}

__global__ void proj_x_kernel(const float* __restrict__ x,
                              const float* __restrict__ ME,
                              const float* __restrict__ Wpt, const float* __restrict__ bpt,
                              const float* __restrict__ Wex, const float* __restrict__ bex,
                              const float* __restrict__ lng, const float* __restrict__ lnb,
                              float*  __restrict__ token,
                              bf16_t* __restrict__ PT,
                              bf16_t* __restrict__ EV) {
  __shared__ float xs[DIN_];
  __shared__ float red[DP_];
  const int row = blockIdx.x;
  const int d   = threadIdx.x;
  if (d < DIN_) xs[d] = x[(size_t)row * DIN_ + d];
  __syncthreads();

  float t = 0.f, pt = 0.f, ev = 0.f;
  if (d < D_) {
#pragma unroll 5
    for (int k = 0; k < DIN_; ++k) {
      const float xv = xs[k];
      t  += xv * ME [k * D_ + d];
      pt += xv * Wpt[k * D_ + d];
      ev += xv * Wex[k * D_ + d];
    }
    pt = tanhf(pt + bpt[d]);
    ev = tanhf(ev + bex[d]);
  }
  PT[(size_t)row * DP_ + d] = (bf16_t)((d < D_) ? pt : 0.f);
  EV[(size_t)row * DP_ + d] = (bf16_t)((d < D_) ? ev : 0.f);

  const float mu  = block_sum_320((d < D_) ? t : 0.f, red, d) * (1.f / (float)D_);
  const float dv  = (d < D_) ? (t - mu) : 0.f;
  const float var = block_sum_320(dv * dv, red, d) * (1.f / (float)D_);
  if (d < D_)
    token[(size_t)row * D_ + d] = lng[d] * (t - mu) * rsqrtf(var + 1e-6f) + lnb[d];
}

// ======================================================================
// Projection of past rows.
// ======================================================================
__global__ void proj_past_kernel(const float* __restrict__ past,
                                 const float* __restrict__ Wpg, const float* __restrict__ bpg,
                                 const float* __restrict__ Wps, const float* __restrict__ bps,
                                 bf16_t* __restrict__ PVT,
                                 bf16_t* __restrict__ PS) {
  const long idx = (long)blockIdx.x * blockDim.x + threadIdx.x;
  if (idx >= (long)B_ * P_ * DP_) return;
  const int row = (int)(idx / DP_);
  const int d   = (int)(idx % DP_);
  float pv = 0.f, ps = 0.f;
  if (d < D_) {
    const float* pr = past + (size_t)row * DIN_;
#pragma unroll 5
    for (int k = 0; k < DIN_; ++k) {
      const float v = pr[k];
      pv += v * Wpg[k * D_ + d];
      ps += v * Wps[k * D_ + d];
    }
    pv = tanhf(pv + bpg[d]);
    ps = tanhf(ps + bps[d]);
  }
  const int b = row / P_, p = row % P_;
  PVT[((size_t)b * DP_ + d) * P_ + p] = (bf16_t)pv;
  PS [(size_t)row * DP_ + d]          = (bf16_t)ps;
}

// ======================================================================
// Weight prep: W1 -> bf16 [320,320] padded, W2 -> bf16 [320,1024] padded.
// ======================================================================
__global__ void prep_w_kernel(const float* __restrict__ W1,
                              const float* __restrict__ W2,
                              bf16_t* __restrict__ W1b,
                              bf16_t* __restrict__ W2b) {
  long idx = (long)blockIdx.x * blockDim.x + threadIdx.x;
  if (idx < (long)DP_ * DP_) {
    const int k = (int)(idx / DP_), n = (int)(idx % DP_);
    W1b[idx] = (bf16_t)((k < D_ && n < D_) ? W1[k * D_ + n] : 0.f);
    return;
  }
  idx -= (long)DP_ * DP_;
  if (idx < (long)DP_ * OUT_) {
    const int k = (int)(idx / OUT_), n = (int)(idx % OUT_);
    W2b[idx] = (bf16_t)((k < D_) ? W2[k * OUT_ + n] : 0.f);
  }
}

// ======================================================================
// Workgroup-tiled batched WMMA GEMM (NN form, bf16 x bf16 -> f32):
//   C[b][m][n] = sum_k A[b][m][k] * B[b][k][n]
// WG = 256 threads (8 waves). WG tile = 128 x NTILE, K-step 32,
// double-buffered LDS, async global->LDS when available.
// Wave layout: 4 (M) x 2 (N); wave tile = 32 x NTILE/2.
// M must be a multiple of 128, N a multiple of NTILE, K of 32.
// ======================================================================
enum { EPI_SIG = 0, EPI_DUAL = 1, EPI_TOKEN = 2, EPI_RELU = 3, EPI_BIAS_F32 = 4 };

template <int EPI, int NTILE>
__global__ __launch_bounds__(256)
void wmma_gemm_kernel(const bf16_t* __restrict__ A, int lda, long strA,
                      const bf16_t* __restrict__ Bm, int ldb, long strB,
                      void* __restrict__ O, int ldo, long strO,
                      bf16_t* __restrict__ O2, int ldo2, long strO2,
                      const float* __restrict__ aux, long strAux,
                      int M, int N, int K, int nb) {
  static_assert(NTILE == 64 || NTILE == 128, "NTILE");
  constexpr int WNT  = NTILE / 2;   // wave N extent
  constexpr int NT16 = WNT / 16;    // 16-wide tiles per wave (2 or 4)
  constexpr int CPR  = NTILE / 8;   // 16B chunks per B row

  __shared__ alignas(64) bf16_t sA[2][128 * 32];
  __shared__ alignas(64) bf16_t sB[2][32 * NTILE];

  const int tid    = threadIdx.x;
  const int lane   = tid & 31;
  const int wid    = tid >> 5;
  const int half   = lane >> 4;
  const int l15    = lane & 15;
  const int tilesN = N / NTILE;
  const int tilesM = M >> 7;

  const int tn = blockIdx.x % tilesN;
  const int tm = (blockIdx.x / tilesN) % tilesM;
  const int b  = blockIdx.x / (tilesN * tilesM);
  const int m0 = tm << 7;
  const int n0 = tn * NTILE;
  const int wm = (wid >> 1) * 32;   // wave M offset in WG tile
  const int wn = (wid & 1) * WNT;   // wave N offset in WG tile

  // ---- per-thread staging assignment (16B chunks) ----
  const int ar = tid >> 2;                 // A row 0..63 (chunk2: +64)
  const int ak = (tid & 3) << 3;           // A k-offset
  const int br = tid / CPR;                // B k-row (chunk2: +16, NTILE==128)
  const int bn = (tid % CPR) << 3;         // B n-offset
  const bf16_t* gA = A  + (long)b * strA + (long)(m0 + ar) * lda + ak;
  const bf16_t* gB = Bm + (long)b * strB + (long)br * ldb + n0 + bn;
  const int lA = ar * 32 + ak;
  const int lB = br * NTILE + bn;

  v8f acc[2][NT16] = {};

  auto compute = [&](int bi) {
    v16bf af[2];
#pragma unroll
    for (int mi = 0; mi < 2; ++mi) {
      const bf16_t* arow = &sA[bi][(wm + mi * 16 + l15) * 32];
      const v8bf alo = *(const v8bf*)(arow + 8 * half);
      const v8bf ahi = *(const v8bf*)(arow + 16 + 8 * half);
#pragma unroll
      for (int j = 0; j < 8; ++j) { af[mi][j] = alo[j]; af[mi][8 + j] = ahi[j]; }
    }
#pragma unroll
    for (int ni = 0; ni < NT16; ++ni) {
      v16bf bfr;
#pragma unroll
      for (int j = 0; j < 8; ++j) {
        bfr[j]     = sB[bi][(8 * half + j) * NTILE + wn + ni * 16 + l15];
        bfr[8 + j] = sB[bi][(16 + 8 * half + j) * NTILE + wn + ni * 16 + l15];
      }
#pragma unroll
      for (int mi = 0; mi < 2; ++mi)
        acc[mi][ni] = __builtin_amdgcn_wmma_f32_16x16x32_bf16(
            false, af[mi], false, bfr, (short)0, acc[mi][ni], false, false);
    }
  };

  const int KT = K >> 5;
  int buf = 0;
#if USE_ASYNC
  auto stage = [&](int bi, int kt) {
    const int k0 = kt << 5;
    cp_async16(gA + k0,                 &sA[bi][lA]);
    cp_async16(gA + k0 + 64 * lda,      &sA[bi][lA + 64 * 32]);
    cp_async16(gB + (long)k0 * ldb,     &sB[bi][lB]);
    if (NTILE == 128)
      cp_async16(gB + (long)(k0 + 16) * ldb, &sB[bi][lB + 16 * NTILE]);
  };
  stage(0, 0);
  __builtin_amdgcn_s_wait_asynccnt(0);
  __syncthreads();
  for (int kt = 0; kt < KT; ++kt) {
    if (kt + 1 < KT) stage(buf ^ 1, kt + 1);
    compute(buf);
    __builtin_amdgcn_s_wait_asynccnt(0);
    __syncthreads();
    buf ^= 1;
  }
#else
  v8bf rA0, rA1, rB0, rB1;
  auto loadRegs = [&](int kt) {
    const int k0 = kt << 5;
    rA0 = *(const v8bf*)(gA + k0);
    rA1 = *(const v8bf*)(gA + k0 + 64 * lda);
    rB0 = *(const v8bf*)(gB + (long)k0 * ldb);
    if (NTILE == 128) rB1 = *(const v8bf*)(gB + (long)(k0 + 16) * ldb);
  };
  auto storeRegs = [&](int bi) {
    *(v8bf*)(&sA[bi][lA]) = rA0;
    *(v8bf*)(&sA[bi][lA + 64 * 32]) = rA1;
    *(v8bf*)(&sB[bi][lB]) = rB0;
    if (NTILE == 128) *(v8bf*)(&sB[bi][lB + 16 * NTILE]) = rB1;
  };
  loadRegs(0); storeRegs(0); __syncthreads();
  for (int kt = 0; kt < KT; ++kt) {
    const bool more = (kt + 1 < KT);
    if (more) loadRegs(kt + 1);
    compute(buf);
    __syncthreads();
    if (more) storeRegs(buf ^ 1);
    __syncthreads();
    buf ^= 1;
  }
#endif

  // ---- epilogue ----
#pragma unroll
  for (int mi = 0; mi < 2; ++mi) {
#pragma unroll
    for (int ni = 0; ni < NT16; ++ni) {
      const int n  = n0 + wn + ni * 16 + l15;
      const int mb = m0 + wm + mi * 16 + 8 * half;
      const v8f a = acc[mi][ni];
      if (EPI == EPI_SIG) {
        bf16_t* o = (bf16_t*)O + (long)b * strO;
#pragma unroll
        for (int r = 0; r < 8; ++r)
          o[(long)(mb + r) * ldo + n] = (bf16_t)(1.f / (1.f + __expf(-a[r])));
      } else if (EPI == EPI_DUAL) {
        bf16_t* o = (bf16_t*)O + (long)b * strO;
        v8bf tv;
#pragma unroll
        for (int r = 0; r < 8; ++r) {
          const bf16_t v = (bf16_t)a[r];
          o[(long)(mb + r) * ldo + n] = v;
          tv[r] = v;
        }
        *(v8bf*)(O2 + (long)b * strO2 + (long)n * ldo2 + mb) = tv;
      } else if (EPI == EPI_TOKEN) {
        bf16_t* o = (bf16_t*)O + (long)b * strO;
        const float* tk = aux + (long)b * strAux;
#pragma unroll
        for (int r = 0; r < 8; ++r) {
          float v = a[r];
          if (n < D_) v += tk[(long)(mb + r) * D_ + n];
          o[(long)(mb + r) * ldo + n] = (bf16_t)v;
        }
      } else if (EPI == EPI_RELU) {
        bf16_t* o = (bf16_t*)O + (long)b * strO;
        const float bias = (n < D_) ? aux[n] : 0.f;
#pragma unroll
        for (int r = 0; r < 8; ++r) {
          float v = a[r] + bias;
          o[(long)(mb + r) * ldo + n] = (bf16_t)(v > 0.f ? v : 0.f);
        }
      } else {  // EPI_BIAS_F32
        float* o = (float*)O + (long)b * strO;
        const float bias = aux[n];
#pragma unroll
        for (int r = 0; r < 8; ++r)
          o[(long)(mb + r) * ldo + n] = a[r] + bias;
      }
    }
  }
}

// ======================================================================
extern "C" void kernel_launch(void* const* d_in, const int* in_sizes, int n_in,
                              void* d_out, int out_size, void* d_ws, size_t ws_size,
                              hipStream_t stream) {
  (void)in_sizes; (void)n_in; (void)out_size; (void)ws_size;
  const float* x    = (const float*)d_in[0];
  const float* ME   = (const float*)d_in[1];
  const float* past = (const float*)d_in[2];
  const float* wpt  = (const float*)d_in[3];
  const float* bpt  = (const float*)d_in[4];
  const float* wpg  = (const float*)d_in[5];
  const float* bpg  = (const float*)d_in[6];
  const float* wps  = (const float*)d_in[7];
  const float* bps  = (const float*)d_in[8];
  const float* wex  = (const float*)d_in[9];
  const float* bex  = (const float*)d_in[10];
  const float* lng  = (const float*)d_in[11];
  const float* lnb  = (const float*)d_in[12];
  const float* W1   = (const float*)d_in[13];
  const float* b1   = (const float*)d_in[14];
  const float* W2   = (const float*)d_in[15];
  const float* b2   = (const float*)d_in[16];
  float* out = (float*)d_out;

  // ---- workspace carve-up (256B aligned) ----
  char*  ws  = (char*)d_ws;
  size_t off = 0;
  auto carve = [&](size_t bytes) { size_t o = off; off += (bytes + 255) & ~(size_t)255; return o; };
  float*  TK  = (float*) (ws + carve((size_t)BS_ * D_  * 4));
  bf16_t* PT  = (bf16_t*)(ws + carve((size_t)BS_ * DP_ * 2));
  bf16_t* EV  = (bf16_t*)(ws + carve((size_t)BS_ * DP_ * 2));
  bf16_t* PVT = (bf16_t*)(ws + carve((size_t)B_ * DP_ * P_ * 2));
  bf16_t* PS  = (bf16_t*)(ws + carve((size_t)B_ * P_ * DP_ * 2));
  bf16_t* G1  = (bf16_t*)(ws + carve((size_t)B_ * S_ * P_ * 2));   // gate buf (reused)
  bf16_t* PRE = (bf16_t*)(ws + carve((size_t)B_ * S_ * DP_ * 2));
  bf16_t* PRT = (bf16_t*)(ws + carve((size_t)B_ * DP_ * S_ * 2));
  bf16_t* W1b = (bf16_t*)(ws + carve((size_t)DP_ * DP_  * 2));
  bf16_t* W2b = (bf16_t*)(ws + carve((size_t)DP_ * OUT_ * 2));
  bf16_t* FT  = PT;   // filter_token reuses PT (dead after gate1)
  bf16_t* H   = EV;   // relu hidden reuses EV (dead after gate2)

  // ---- 1. projections + weight prep ----
  proj_x_kernel<<<BS_, DP_, 0, stream>>>(x, ME, wpt, bpt, wex, bex, lng, lnb, TK, PT, EV);
  {
    const long n = (long)B_ * P_ * DP_;
    proj_past_kernel<<<(unsigned)((n + 255) / 256), 256, 0, stream>>>(past, wpg, bpg, wps, bps, PVT, PS);
  }
  {
    const long n = (long)DP_ * DP_ + (long)DP_ * OUT_;
    prep_w_kernel<<<(unsigned)((n + 255) / 256), 256, 0, stream>>>(W1, W2, W1b, W2b);
  }

  auto nblk = [](long M, long N, long NT, long nb) { return (unsigned)(nb * (M / 128) * (N / NT)); };

  // ---- 2. gate1 = sigmoid(past_token @ past_vector^T)  [B,S,P] ----
  wmma_gemm_kernel<EPI_SIG, 128><<<nblk(S_, P_, 128, B_), 256, 0, stream>>>(
      PT, DP_, (long)S_ * DP_, PVT, P_, (long)DP_ * P_,
      G1, P_, (long)S_ * P_, nullptr, 0, 0, nullptr, 0,
      S_, P_, DP_, B_);

  // ---- 3. pre_state = gate1 @ past_state (+ transposed copy) ----
  wmma_gemm_kernel<EPI_DUAL, 64><<<nblk(S_, DP_, 64, B_), 256, 0, stream>>>(
      G1, P_, (long)S_ * P_, PS, DP_, (long)P_ * DP_,
      PRE, DP_, (long)S_ * DP_, PRT, S_, (long)DP_ * S_, nullptr, 0,
      S_, DP_, P_, B_);

  // ---- 4. gate2 = sigmoid(expose_vector @ pre_state^T)  [B,S,S] ----
  wmma_gemm_kernel<EPI_SIG, 128><<<nblk(S_, S_, 128, B_), 256, 0, stream>>>(
      EV, DP_, (long)S_ * DP_, PRT, S_, (long)DP_ * S_,
      G1, S_, (long)S_ * S_, nullptr, 0, 0, nullptr, 0,
      S_, S_, DP_, B_);

  // ---- 5. filter = token + gate2 @ pre_state ----
  wmma_gemm_kernel<EPI_TOKEN, 64><<<nblk(S_, DP_, 64, B_), 256, 0, stream>>>(
      G1, S_, (long)S_ * S_, PRE, DP_, (long)S_ * DP_,
      FT, DP_, (long)S_ * DP_, nullptr, 0, 0, TK, (long)S_ * D_,
      S_, DP_, S_, B_);

  // ---- 6. h = relu(filter @ W1 + b1) ----
  wmma_gemm_kernel<EPI_RELU, 64><<<nblk(BS_, DP_, 64, 1), 256, 0, stream>>>(
      FT, DP_, 0, W1b, DP_, 0,
      H, DP_, 0, nullptr, 0, 0, b1, 0,
      BS_, DP_, DP_, 1);

  // ---- 7. out = h @ W2 + b2 ----
  wmma_gemm_kernel<EPI_BIAS_F32, 128><<<nblk(BS_, OUT_, 128, 1), 256, 0, stream>>>(
      H, DP_, 0, W2b, OUT_, 0,
      out, OUT_, 0, nullptr, 0, 0, b2, 0,
      BS_, OUT_, DP_, 1);
}
